// SelectiveCrossSSM_16114717294733
// MI455X (gfx1250) — matrix operand order repaired and checked by reading, not compiled, not run on hardware
//
#include <hip/hip_runtime.h>
#include <hip/hip_bf16.h>
#include <math.h>
#include <stdint.h>

// ---------------------------------------------------------------------------
// Problem constants (from reference)
// ---------------------------------------------------------------------------
#define BATCH   4
#define SEQ     1024
#define DMODEL  512
#define DINNER  1024
#define NSTATE  16
#define NHEADS  8
#define DHEAD   64
#define ROWS    (BATCH * SEQ)      // 4096 token rows

typedef __bf16 bf16;
typedef __attribute__((ext_vector_type(16))) __bf16 v16bf;
typedef __attribute__((ext_vector_type(8)))  float  v8f;

#define V8F_ZERO (v8f{0.0f, 0.0f, 0.0f, 0.0f, 0.0f, 0.0f, 0.0f, 0.0f})

union FragBF { v16bf v; uint4 u[2]; };

enum { ACT_NONE = 0, ACT_GELU = 1, ACT_SILU = 2, ACT_SOFTPLUS = 3, ACT_SIGMOID = 4 };

// ---- CDNA5 async global->LDS path (guarded; falls back to load+ds_store) ----
#if defined(__has_builtin)
#  if __has_builtin(__builtin_amdgcn_global_load_async_to_lds_b128) && \
      __has_builtin(__builtin_amdgcn_s_wait_asynccnt)
#    define HAVE_ASYNC_LDS 1
#  endif
#endif
#ifndef HAVE_ASYNC_LDS
#  define HAVE_ASYNC_LDS 0
#endif

#if HAVE_ASYNC_LDS
// Builtin signature (from hipcc diagnostics): (AS1 v4i* src, AS3 v4i* dst, Ii, Ii).
// Low 32 bits of a generic LDS address are the wave-relative LDS byte offset
// (ISA 10.2 aperture mapping), so inttoptr to AS(3) is the LDS offset.
typedef __attribute__((__vector_size__(4 * sizeof(int)))) int v4i_t;
#define TO_LDS(p) ((__attribute__((address_space(3))) v4i_t*)(uint32_t)(uintptr_t)(p))
#define TO_GLB(p) ((__attribute__((address_space(1))) v4i_t*)(uintptr_t)(p))
#define ASYNC_CP16(dst_lds, src_glb) \
    __builtin_amdgcn_global_load_async_to_lds_b128(TO_GLB(src_glb), TO_LDS(dst_lds), 0, 0)
#define ASYNC_WAIT() __builtin_amdgcn_s_wait_asynccnt(0)
#endif

__device__ __forceinline__ float act_apply(int act, float x) {
    switch (act) {
        case ACT_GELU:     return 0.5f * x * (1.0f + erff(x * 0.70710678118654752f));
        case ACT_SILU:     return x / (1.0f + __expf(-x));
        case ACT_SOFTPLUS: return (x > 20.0f) ? x : log1pf(__expf(x));
        case ACT_SIGMOID:  return 1.0f / (1.0f + __expf(-x));
        default:           return x;
    }
}

// ---------------------------------------------------------------------------
// f32 -> bf16 convert
// ---------------------------------------------------------------------------
__global__ __launch_bounds__(256) void cvt_f32_bf16_kernel(const float* __restrict__ in,
                                                           bf16* __restrict__ out, int n) {
    int i = blockIdx.x * 256 + threadIdx.x;
    if (i < n) out[i] = (bf16)in[i];
}

// ---------------------------------------------------------------------------
// Generic bf16 WMMA GEMM:  C = act(A[MxK] * W[KxN] + bias) (+ resid)
//   Block tile 128(M) x 128(N), K step 32, double-buffered LDS. 8 waves,
//   each wave a 32x64 tile (2 A-frags x 4 B-frags -> 8 wmma per K step).
//   Tiles staged via async global->LDS (ASYNCcnt) when available; the async
//   copy of tile t+1 overlaps the WMMA compute of tile t.
//   Accumulators are plain v8f SSA values (no union) so they live in VGPRs.
// Requirements: M % 128 == 0, K % 32 == 0, N % 8 == 0 (N<128 handled by guard).
// ---------------------------------------------------------------------------
__global__ __launch_bounds__(256, 1)
void gemm_bf16_kernel(const bf16* __restrict__ A, int lda,
                      const bf16* __restrict__ W, int ldw,
                      const float* __restrict__ bias,
                      float* __restrict__ Cf, int ldc,
                      bf16* __restrict__ Cb, int ldcb,
                      const float* __restrict__ resid, int ldr,
                      int M, int N, int K, int act) {
    __shared__ __align__(16) bf16 As[2][128][40];    // 32-wide K tile, padded stride
    __shared__ __align__(16) bf16 Bs[2][32][136];    // 128-wide N tile, padded stride

    const int tid  = threadIdx.x;
    const int wv   = tid >> 5;
    const int lane = tid & 31;
    const int mw   = wv >> 1;        // 0..3 : wave row (32 rows each)
    const int nw   = wv & 1;         // 0..1 : wave col (64 cols each)
    const int hl   = lane >> 4;      // lane half (K-half selector for A frags)
    const int l15  = lane & 15;
    const int m0   = blockIdx.y * 128;
    const int n0   = blockIdx.x * 128;
    const bool bfull = (n0 + 128 <= N);

    v8f acc[2][4];
#pragma unroll
    for (int i = 0; i < 2; i++)
#pragma unroll
        for (int j = 0; j < 4; j++) acc[i][j] = V8F_ZERO;

    const int arow = tid >> 1, acol = (tid & 1) * 16;   // A tile: 16 elems/thread
    const int brow = tid >> 3, bcol = (tid & 7) * 16;   // B tile: 16 elems/thread

    auto stage_tile = [&](int t, int buf) {
        const int kt = t * 32;
        const bf16* ag = A + (size_t)(m0 + arow) * lda + kt + acol;
        const bf16* wg = W + (size_t)(kt + brow) * ldw + n0 + bcol;
#if HAVE_ASYNC_LDS
        ASYNC_CP16(&As[buf][arow][acol],     ag);
        ASYNC_CP16(&As[buf][arow][acol + 8], ag + 8);
        if (bfull) {
            ASYNC_CP16(&Bs[buf][brow][bcol],     wg);
            ASYNC_CP16(&Bs[buf][brow][bcol + 8], wg + 8);
        } else {
            uint4 b0 = {0, 0, 0, 0}, b1 = {0, 0, 0, 0};
            if (n0 + bcol < N)     b0 = *(const uint4*)wg;
            if (n0 + bcol + 8 < N) b1 = *(const uint4*)(wg + 8);
            *(uint4*)&Bs[buf][brow][bcol]     = b0;
            *(uint4*)&Bs[buf][brow][bcol + 8] = b1;
        }
#else
        *(uint4*)&As[buf][arow][acol]     = *(const uint4*)ag;
        *(uint4*)&As[buf][arow][acol + 8] = *(const uint4*)(ag + 8);
        uint4 b0 = {0, 0, 0, 0}, b1 = {0, 0, 0, 0};
        if (n0 + bcol < N)     b0 = *(const uint4*)wg;
        if (n0 + bcol + 8 < N) b1 = *(const uint4*)(wg + 8);
        *(uint4*)&Bs[buf][brow][bcol]     = b0;
        *(uint4*)&Bs[buf][brow][bcol + 8] = b1;
#endif
        __builtin_prefetch(ag + 64, 0, 3);               // tile t+2 (WGP scope)
        __builtin_prefetch(wg + (size_t)64 * ldw, 0, 3);
    };

    const int nkt = K >> 5;
    stage_tile(0, 0);
#if HAVE_ASYNC_LDS
    ASYNC_WAIT();
#endif
    __syncthreads();

    for (int t = 0; t < nkt; ++t) {
        const int cur = t & 1;
        if (t + 1 < nkt) stage_tile(t + 1, cur ^ 1);   // overlap with compute

        // --- build fragments (ISA 16-bit A / B VGPR layouts) ---
        FragBF a[2], b[4];
#pragma unroll
        for (int i = 0; i < 2; i++) {
            const bf16* p = &As[cur][mw * 32 + i * 16 + l15][hl * 8];
            a[i].u[0] = *(const uint4*)p;          // K = hl*8 + 0..7
            a[i].u[1] = *(const uint4*)(p + 16);   // K = 16 + hl*8 + 0..7
        }
#pragma unroll
        for (int j = 0; j < 4; j++) {
            const bf16* p = &Bs[cur][0][nw * 64 + j * 16 + l15];
#pragma unroll
            for (int s = 0; s < 16; s++)
                b[j].v[s] = p[((s >> 3) * 16 + hl * 8 + (s & 7)) * 136];
        }
#pragma unroll
        for (int i = 0; i < 2; i++)
#pragma unroll
            for (int j = 0; j < 4; j++)
                acc[i][j] = __builtin_amdgcn_wmma_f32_16x16x32_bf16(
                    false, a[i].v, false, b[j].v, (short)0, acc[i][j], false, false);

#if HAVE_ASYNC_LDS
        ASYNC_WAIT();          // next tile resident before anyone reads it
#endif
        __syncthreads();
    }

    // --- epilogue: bias + activation + optional residual, dual-precision store ---
#pragma unroll
    for (int i = 0; i < 2; i++)
#pragma unroll
        for (int j = 0; j < 4; j++)
#pragma unroll
            for (int r = 0; r < 8; r++) {
                const int gm = m0 + mw * 32 + i * 16 + r + hl * 8;
                const int gn = n0 + nw * 64 + j * 16 + l15;
                if (gn < N) {
                    float v = acc[i][j][r];
                    if (bias)  v += bias[gn];
                    v = act_apply(act, v);
                    if (resid) v += resid[(size_t)gm * ldr + gn];
                    if (Cf) Cf[(size_t)gm * ldc + gn] = v;
                    if (Cb) Cb[(size_t)gm * ldcb + gn] = (bf16)v;
                }
            }
}

// ---------------------------------------------------------------------------
// Depthwise causal conv (K=4) + bias + SiLU.
// Reads bx = bxz[..., :DINNER]; writes f32 (for scan) and bf16 into the
// [bx | ctx] concat buffer (stride 1536).
// ---------------------------------------------------------------------------
__global__ __launch_bounds__(256)
void conv_silu_kernel(const float* __restrict__ bxz, const float* __restrict__ convW,
                      const float* __restrict__ convb,
                      float* __restrict__ bx_silu, bf16* __restrict__ catb) {
    const int idx = blockIdx.x * 256 + threadIdx.x;     // over ROWS*DINNER
    const int c   = idx & (DINNER - 1);
    const int row = idx >> 10;
    const int t   = row & (SEQ - 1);
    const int rowb = row - t;                           // b*SEQ
    float acc = convb[c];
#pragma unroll
    for (int k = 0; k < 4; k++) {
        const int tp = t - 3 + k;
        if (tp >= 0) acc += convW[c * 4 + k] * bxz[(size_t)(rowb + tp) * (2 * DINNER) + c];
    }
    const float s = acc / (1.0f + __expf(-acc));
    bx_silu[(size_t)row * DINNER + c] = s;
    catb[(size_t)row * (DINNER + DMODEL) + c] = (bf16)s;
}

// ---------------------------------------------------------------------------
// Selective SSM scan. One thread per (b, d) channel, N=16 state in registers.
// B_t / C_t (32 floats, shared per batch-timestep) broadcast through LDS.
// Fuses +D*x and *silu(z); writes bf16 for the ssm_out GEMM.
// ---------------------------------------------------------------------------
__global__ __launch_bounds__(256)
void ssm_scan_kernel(const float* __restrict__ dt, const float* __restrict__ bc,
                     const float* __restrict__ x, const float* __restrict__ bxz,
                     const float* __restrict__ A_log, const float* __restrict__ D_param,
                     bf16* __restrict__ yb) {
    const int b = blockIdx.x >> 2;
    const int d = (blockIdx.x & 3) * 256 + threadIdx.x;
    float a[NSTATE], h[NSTATE];
#pragma unroll
    for (int n = 0; n < NSTATE; n++) { a[n] = -__expf(A_log[d * NSTATE + n]); h[n] = 0.0f; }
    const float Dp = D_param[d];
    __shared__ float sBC[2 * NSTATE];
    for (int t = 0; t < SEQ; t++) {
        const size_t row = (size_t)b * SEQ + t;
        if (threadIdx.x < 2 * NSTATE) sBC[threadIdx.x] = bc[row * (2 * NSTATE) + threadIdx.x];
        __syncthreads();
        const float dtv = dt[row * DINNER + d];
        const float xv  = x[row * DINNER + d];
        const float dtx = dtv * xv;
        float acc = 0.0f;
#pragma unroll
        for (int n = 0; n < NSTATE; n++) {
            h[n] = __expf(dtv * a[n]) * h[n] + dtx * sBC[n];
            acc += h[n] * sBC[NSTATE + n];
        }
        float y = acc + Dp * xv;
        const float z = bxz[row * (2 * DINNER) + DINNER + d];   // bz slice
        y *= z / (1.0f + __expf(-z));
        yb[row * DINNER + d] = (bf16)y;
        __syncthreads();
    }
}

// ---------------------------------------------------------------------------
// LayerNorm (width 512), one wave per row, writes bf16.
// ---------------------------------------------------------------------------
__global__ __launch_bounds__(256)
void layernorm_bf16_kernel(const float* __restrict__ X, const float* __restrict__ g,
                           const float* __restrict__ beta, bf16* __restrict__ Yb) {
    const int wv = threadIdx.x >> 5, lane = threadIdx.x & 31;
    const int row = blockIdx.x * 8 + wv;
    const float* x = X + (size_t)row * DMODEL;
    float v[16];
    float sum = 0.0f;
#pragma unroll
    for (int j = 0; j < 16; j++) { v[j] = x[lane + j * 32]; sum += v[j]; }
#pragma unroll
    for (int d = 1; d < 32; d <<= 1) sum += __shfl_xor(sum, d, 32);
    const float mean = sum * (1.0f / DMODEL);
    float s2 = 0.0f;
#pragma unroll
    for (int j = 0; j < 16; j++) { const float t = v[j] - mean; s2 += t * t; }
#pragma unroll
    for (int d = 1; d < 32; d <<= 1) s2 += __shfl_xor(s2, d, 32);
    const float rstd = rsqrtf(s2 * (1.0f / DMODEL) + 1e-5f);
#pragma unroll
    for (int j = 0; j < 16; j++) {
        const int c = lane + j * 32;
        Yb[(size_t)row * DMODEL + c] = (bf16)((v[j] - mean) * rstd * g[c] + beta[c]);
    }
}

// ---------------------------------------------------------------------------
// Flash attention. Block = (b, h, 128-query tile); 8 waves x 16 q-rows.
// dh = 64. Scores & PV via bf16 WMMA; online softmax with 16-lane
// half-wave shuffle reductions (wave32 C-fragment row layout).
// Q/V staged via async global->LDS when available. Accumulators are plain
// v8f SSA values so they stay in VGPRs.
// ---------------------------------------------------------------------------
__global__ __launch_bounds__(256, 1)
void attention_kernel(const bf16* __restrict__ Qp,   // [ROWS,512]
                      const bf16* __restrict__ KVp,  // [ROWS,1024] = [K|V]
                      bf16* __restrict__ Ob) {       // [ROWS,512]
    __shared__ __align__(16) bf16 Qs[128][72];
    __shared__ __align__(16) bf16 Kst[64][72];       // transposed: [c][key]
    __shared__ __align__(16) bf16 Vs[64][72];        // [key][c]
    __shared__ __align__(16) bf16 Ps[8][16][72];     // per-wave P staging

    const int tid = threadIdx.x, wv = tid >> 5, lane = tid & 31;
    const int hl = lane >> 4, l15 = lane & 15;
    const int bh = blockIdx.y, b = bh >> 3, h = bh & 7;
    const int q0 = blockIdx.x * 128;

    // --- stage Q tile once ---
    {
        const int row = tid >> 1, c0 = (tid & 1) * 32;
        const bf16* gq = Qp + (size_t)(b * SEQ + q0 + row) * DMODEL + h * DHEAD + c0;
#if HAVE_ASYNC_LDS
        ASYNC_CP16(&Qs[row][c0],      gq);
        ASYNC_CP16(&Qs[row][c0 + 8],  gq + 8);
        ASYNC_CP16(&Qs[row][c0 + 16], gq + 16);
        ASYNC_CP16(&Qs[row][c0 + 24], gq + 24);
        ASYNC_WAIT();
#else
        *(uint4*)&Qs[row][c0]      = *(const uint4*)gq;
        *(uint4*)&Qs[row][c0 + 8]  = *(const uint4*)(gq + 8);
        *(uint4*)&Qs[row][c0 + 16] = *(const uint4*)(gq + 16);
        *(uint4*)&Qs[row][c0 + 24] = *(const uint4*)(gq + 24);
#endif
    }
    __syncthreads();
    FragBF aq[2];
#pragma unroll
    for (int cb = 0; cb < 2; cb++) {
        const bf16* p = &Qs[wv * 16 + l15][cb * 32 + hl * 8];
        aq[cb].u[0] = *(const uint4*)p;
        aq[cb].u[1] = *(const uint4*)(p + 16);
    }

    v8f o[4];
#pragma unroll
    for (int jt = 0; jt < 4; jt++) o[jt] = V8F_ZERO;
    float mrow[8], srow[8];
#pragma unroll
    for (int r = 0; r < 8; r++) { mrow[r] = -1e30f; srow[r] = 0.0f; }

    const int lk = tid >> 2, lc = (tid & 3) * 16;    // K/V tile loader coords
    for (int kt = 0; kt < SEQ; kt += 64) {
        __syncthreads();
        {   // K transposed (regs) + V staged (async)
            const bf16* gk = KVp + (size_t)(b * SEQ + kt + lk) * (2 * DMODEL) + h * DHEAD + lc;
            const bf16* gv = gk + DMODEL;            // V half of KV buffer
#if HAVE_ASYNC_LDS
            ASYNC_CP16(&Vs[lk][lc],     gv);
            ASYNC_CP16(&Vs[lk][lc + 8], gv + 8);
#else
            *(uint4*)&Vs[lk][lc]     = *(const uint4*)gv;
            *(uint4*)&Vs[lk][lc + 8] = *(const uint4*)(gv + 8);
#endif
            bf16 tmp[16];
            *(uint4*)&tmp[0] = *(const uint4*)gk;
            *(uint4*)&tmp[8] = *(const uint4*)(gk + 8);
#pragma unroll
            for (int e = 0; e < 16; e++) Kst[lc + e][lk] = tmp[e];
#if HAVE_ASYNC_LDS
            ASYNC_WAIT();
#endif
        }
        __syncthreads();

        // --- S = (Q K^T) * scale ---
        v8f s[4];
#pragma unroll
        for (int jt = 0; jt < 4; jt++) s[jt] = V8F_ZERO;
#pragma unroll
        for (int cb = 0; cb < 2; cb++)
#pragma unroll
            for (int jt = 0; jt < 4; jt++) {
                FragBF bk;
                const bf16* p = &Kst[cb * 32][jt * 16 + l15];
#pragma unroll
                for (int s = 0; s < 16; s++)
                    bk.v[s] = p[((s >> 3) * 16 + hl * 8 + (s & 7)) * 72];
                s[jt] = __builtin_amdgcn_wmma_f32_16x16x32_bf16(
                    false, aq[cb].v, false, bk.v, (short)0, s[jt], false, false);
            }

        // --- online softmax update ---
#pragma unroll
        for (int r = 0; r < 8; r++) {
            float sv[4];
            float mx = -1e30f;
#pragma unroll
            for (int jt = 0; jt < 4; jt++) {
                sv[jt] = s[jt][r] * 0.125f;          // 1/sqrt(64)
                mx = fmaxf(mx, sv[jt]);
            }
#pragma unroll
            for (int d = 1; d < 16; d <<= 1) mx = fmaxf(mx, __shfl_xor(mx, d, 32));
            const float mn = fmaxf(mrow[r], mx);
            const float corr = __expf(mrow[r] - mn);
            mrow[r] = mn;
            float rs = 0.0f;
#pragma unroll
            for (int jt = 0; jt < 4; jt++) {
                const float p = __expf(sv[jt] - mn);
                rs += p;
                Ps[wv][r + hl * 8][jt * 16 + l15] = (bf16)p;
                o[jt][r] = o[jt][r] * corr;
            }
#pragma unroll
            for (int d = 1; d < 16; d <<= 1) rs += __shfl_xor(rs, d, 32);
            srow[r] = srow[r] * corr + rs;
        }
        __syncthreads();

        // --- O += P V ---
#pragma unroll
        for (int kb = 0; kb < 2; kb++) {
            FragBF ap;
            const bf16* p = &Ps[wv][l15][kb * 32 + hl * 8];
            ap.u[0] = *(const uint4*)p;
            ap.u[1] = *(const uint4*)(p + 16);
#pragma unroll
            for (int jt = 0; jt < 4; jt++) {
                FragBF bv;
                const bf16* q = &Vs[kb * 32][jt * 16 + l15];
#pragma unroll
                for (int s = 0; s < 16; s++)
                    bv.v[s] = q[((s >> 3) * 16 + hl * 8 + (s & 7)) * 72];
                o[jt] = __builtin_amdgcn_wmma_f32_16x16x32_bf16(
                    false, ap.v, false, bv.v, (short)0, o[jt], false, false);
            }
        }
    }

    // --- normalize and store bf16 ---
#pragma unroll
    for (int jt = 0; jt < 4; jt++)
#pragma unroll
        for (int r = 0; r < 8; r++) {
            const int row = b * SEQ + q0 + wv * 16 + r + hl * 8;
            const int col = h * DHEAD + jt * 16 + l15;
            Ob[(size_t)row * DMODEL + col] = (bf16)(o[jt][r] / srow[r]);
        }
}

// ---------------------------------------------------------------------------
// h_c = h_audio + g*a_out + (1-g)*bio_f
// ---------------------------------------------------------------------------
__global__ __launch_bounds__(256)
void combine_kernel(const float* __restrict__ ha, const float* __restrict__ g,
                    const float* __restrict__ ao, const float* __restrict__ biof,
                    float* __restrict__ hc) {
    const int i = blockIdx.x * 256 + threadIdx.x;
    const float gv = g[i];
    hc[i] = ha[i] + gv * ao[i] + (1.0f - gv) * biof[i];
}

// ---------------------------------------------------------------------------
// Host launch
// ---------------------------------------------------------------------------
extern "C" void kernel_launch(void* const* d_in, const int* in_sizes, int n_in,
                              void* d_out, int out_size, void* d_ws, size_t ws_size,
                              hipStream_t stream) {
    const float* h_audio   = (const float*)d_in[0];
    const float* h_bio     = (const float*)d_in[1];
    const float* ctx_W     = (const float*)d_in[3];
    const float* ctx_b     = (const float*)d_in[4];
    const float* bio_in_W  = (const float*)d_in[5];
    const float* conv_W    = (const float*)d_in[6];
    const float* conv_b    = (const float*)d_in[7];
    const float* dt_W      = (const float*)d_in[8];
    const float* dt_b      = (const float*)d_in[9];
    const float* bc_W      = (const float*)d_in[10];
    const float* A_log     = (const float*)d_in[11];
    const float* D_param   = (const float*)d_in[12];
    const float* ssm_out_W = (const float*)d_in[13];
    const float* in_proj_W = (const float*)d_in[14];
    const float* in_proj_b = (const float*)d_in[15];
    const float* out_proj_W = (const float*)d_in[16];
    const float* out_proj_b = (const float*)d_in[17];
    const float* nq_g      = (const float*)d_in[18];
    const float* nq_b      = (const float*)d_in[19];
    const float* nkv_g     = (const float*)d_in[20];
    const float* nkv_b     = (const float*)d_in[21];
    const float* gate_W    = (const float*)d_in[22];
    const float* gate_b    = (const float*)d_in[23];
    const float* ffn_ln_g  = (const float*)d_in[24];
    const float* ffn_ln_b  = (const float*)d_in[25];
    const float* ffn_W1    = (const float*)d_in[26];
    const float* ffn_b1    = (const float*)d_in[27];
    const float* ffn_W2    = (const float*)d_in[28];
    const float* ffn_b2    = (const float*)d_in[29];
    float* out = (float*)d_out;

    // ---- workspace carving ----
    char* wsb = (char*)d_ws;
    size_t off = 0;
    auto carve = [&](size_t bytes) -> void* {
        void* p = wsb + off;
        off += (bytes + 255) & ~(size_t)255;
        return p;
    };
    // bf16 weight copies
    bf16* w_ctx = (bf16*)carve((size_t)512 * 512 * 2);
    bf16* w_bio = (bf16*)carve((size_t)512 * 2048 * 2);
    bf16* w_dt  = (bf16*)carve((size_t)1536 * 1024 * 2);
    bf16* w_bc  = (bf16*)carve((size_t)1024 * 32 * 2);
    bf16* w_ssm = (bf16*)carve((size_t)1024 * 512 * 2);
    bf16* w_inp = (bf16*)carve((size_t)512 * 1536 * 2);
    bf16* w_out = (bf16*)carve((size_t)512 * 512 * 2);
    bf16* w_gate = (bf16*)carve((size_t)1024 * 512 * 2);
    bf16* w_f1  = (bf16*)carve((size_t)512 * 1024 * 2);
    bf16* w_f2  = (bf16*)carve((size_t)1024 * 512 * 2);
    // bf16 activations
    bf16* ha_bf   = (bf16*)carve((size_t)ROWS * DMODEL * 2);
    bf16* hb_bf   = (bf16*)carve((size_t)ROWS * DMODEL * 2);
    bf16* cat_bf  = (bf16*)carve((size_t)ROWS * 1536 * 2);   // [silu(conv(bx)) | gelu(ctx)]
    bf16* ssm_y_bf = (bf16*)carve((size_t)ROWS * DINNER * 2);
    bf16* q_bf    = (bf16*)carve((size_t)ROWS * DMODEL * 2);
    bf16* kv_bf   = (bf16*)carve((size_t)ROWS * DMODEL * 2);
    bf16* qp_bf   = (bf16*)carve((size_t)ROWS * DMODEL * 2);
    bf16* kvp_bf  = (bf16*)carve((size_t)ROWS * 1024 * 2);   // [kp | vp]
    bf16* att_bf  = (bf16*)carve((size_t)ROWS * DMODEL * 2);
    bf16* gcat_bf = (bf16*)carve((size_t)ROWS * 1024 * 2);   // [bio_f | a_out]
    bf16* ffn_in_bf = (bf16*)carve((size_t)ROWS * DMODEL * 2);
    bf16* y1_bf   = (bf16*)carve((size_t)ROWS * 1024 * 2);
    // f32 activations
    float* bxz_f   = (float*)carve((size_t)ROWS * 2048 * 4);
    float* bxsilu_f = (float*)carve((size_t)ROWS * DINNER * 4);
    float* dt_f    = (float*)carve((size_t)ROWS * DINNER * 4);
    float* bc_f    = (float*)carve((size_t)ROWS * 32 * 4);
    float* biof_f  = (float*)carve((size_t)ROWS * DMODEL * 4);
    float* aout_f  = (float*)carve((size_t)ROWS * DMODEL * 4);
    float* gate_f  = (float*)carve((size_t)ROWS * DMODEL * 4);
    float* hc_f    = (float*)carve((size_t)ROWS * DMODEL * 4);
    (void)ws_size; (void)n_in; (void)in_sizes; (void)out_size;

    auto cvt = [&](const float* src, bf16* dst, int n) {
        cvt_f32_bf16_kernel<<<(n + 255) / 256, 256, 0, stream>>>(src, dst, n);
    };
    auto gemm = [&](const bf16* A, int lda, const bf16* W, int ldw, const float* bias,
                    float* Cf, int ldc, bf16* Cb, int ldcb,
                    const float* resid, int ldr, int M, int N, int K, int act) {
        dim3 grid((N + 127) / 128, M / 128);
        gemm_bf16_kernel<<<grid, 256, 0, stream>>>(A, lda, W, ldw, bias, Cf, ldc,
                                                   Cb, ldcb, resid, ldr, M, N, K, act);
    };

    // ---- 0) precision conversion ----
    cvt(ctx_W, w_ctx, 512 * 512);
    cvt(bio_in_W, w_bio, 512 * 2048);
    cvt(dt_W, w_dt, 1536 * 1024);
    cvt(bc_W, w_bc, 1024 * 32);
    cvt(ssm_out_W, w_ssm, 1024 * 512);
    cvt(in_proj_W, w_inp, 512 * 1536);
    cvt(out_proj_W, w_out, 512 * 512);
    cvt(gate_W, w_gate, 1024 * 512);
    cvt(ffn_W1, w_f1, 512 * 1024);
    cvt(ffn_W2, w_f2, 1024 * 512);
    cvt(h_audio, ha_bf, ROWS * DMODEL);
    cvt(h_bio, hb_bf, ROWS * DMODEL);

    // ---- 1) ctx = gelu(h_audio @ ctx_W + b) -> bf16 into concat[:,1024:1536] ----
    gemm(ha_bf, 512, w_ctx, 512, ctx_b, nullptr, 0, cat_bf + 1024, 1536,
         nullptr, 0, ROWS, 512, 512, ACT_GELU);
    // ---- 2) bxz = h_bio @ bio_in_W (N=2048, f32) ----
    gemm(hb_bf, 512, w_bio, 2048, nullptr, bxz_f, 2048, nullptr, 0,
         nullptr, 0, ROWS, 2048, 512, ACT_NONE);
    // ---- 3) depthwise causal conv + silu -> f32 + concat[:,0:1024] ----
    conv_silu_kernel<<<(ROWS * DINNER) / 256, 256, 0, stream>>>(bxz_f, conv_W, conv_b,
                                                               bxsilu_f, cat_bf);
    // ---- 4) dt = softplus([bx|ctx] @ dt_W + b) ----
    gemm(cat_bf, 1536, w_dt, 1024, dt_b, dt_f, 1024, nullptr, 0,
         nullptr, 0, ROWS, 1024, 1536, ACT_SOFTPLUS);
    // ---- 5) bc = bx @ bc_W (N=32) ----
    gemm(cat_bf, 1536, w_bc, 32, nullptr, bc_f, 32, nullptr, 0,
         nullptr, 0, ROWS, 32, 1024, ACT_NONE);
    // ---- 6) selective scan (+D*x, *silu(bz)) -> bf16 ----
    ssm_scan_kernel<<<BATCH * (DINNER / 256), 256, 0, stream>>>(dt_f, bc_f, bxsilu_f,
                                                                bxz_f, A_log, D_param,
                                                                ssm_y_bf);
    // ---- 7) bio_f = scan_out @ ssm_out_W -> f32 + bf16 into gate concat[:, :512] ----
    gemm(ssm_y_bf, 1024, w_ssm, 512, nullptr, biof_f, 512, gcat_bf, 1024,
         nullptr, 0, ROWS, 512, 1024, ACT_NONE);
    // ---- 8) layernorms for attention ----
    layernorm_bf16_kernel<<<ROWS / 8, 256, 0, stream>>>(h_audio, nq_g, nq_b, q_bf);
    layernorm_bf16_kernel<<<ROWS / 8, 256, 0, stream>>>(biof_f, nkv_g, nkv_b, kv_bf);
    // ---- 9) projections: qp = q @ W[:, :512]; [kp|vp] = kv @ W[:, 512:] ----
    gemm(q_bf, 512, w_inp, 1536, in_proj_b, nullptr, 0, qp_bf, 512,
         nullptr, 0, ROWS, 512, 512, ACT_NONE);
    gemm(kv_bf, 512, w_inp + 512, 1536, in_proj_b + 512, nullptr, 0, kvp_bf, 1024,
         nullptr, 0, ROWS, 1024, 512, ACT_NONE);
    // ---- 10) flash attention ----
    {
        dim3 grid(SEQ / 128, BATCH * NHEADS);
        attention_kernel<<<grid, 256, 0, stream>>>(qp_bf, kvp_bf, att_bf);
    }
    // ---- 11) a_out = att @ out_proj_W + b -> f32 + bf16 into gate concat[:, 512:] ----
    gemm(att_bf, 512, w_out, 512, out_proj_b, aout_f, 512, gcat_bf + 512, 1024,
         nullptr, 0, ROWS, 512, 512, ACT_NONE);
    // ---- 12) g = sigmoid([bio_f|a_out] @ gate_W + b) ----
    gemm(gcat_bf, 1024, w_gate, 512, gate_b, gate_f, 512, nullptr, 0,
         nullptr, 0, ROWS, 512, 1024, ACT_SIGMOID);
    // ---- 13) h_c = h_audio + g*a_out + (1-g)*bio_f ----
    combine_kernel<<<(ROWS * DMODEL) / 256, 256, 0, stream>>>(h_audio, gate_f, aout_f,
                                                              biof_f, hc_f);
    // ---- 14) FFN: ln -> gelu GEMM -> GEMM (+h_c residual) -> d_out ----
    layernorm_bf16_kernel<<<ROWS / 8, 256, 0, stream>>>(hc_f, ffn_ln_g, ffn_ln_b, ffn_in_bf);
    gemm(ffn_in_bf, 512, w_f1, 1024, ffn_b1, nullptr, 0, y1_bf, 1024,
         nullptr, 0, ROWS, 1024, 512, ACT_GELU);
    gemm(y1_bf, 1024, w_f2, 512, ffn_b2, out, 512, nullptr, 0,
         hc_f, 512, ROWS, 512, 1024, ACT_NONE);
}
